// BlockSparseMoE_79336635892595
// MI455X (gfx1250) — compile-verified, matrix operand order checked by reading, not run on hardware
//
#include <hip/hip_runtime.h>
#include <hip/hip_bf16.h>

// ---- problem dims (compile-time constants from the reference) ----
#define M_TOK 2048
#define HID   1024
#define NEXP  64
#define NMOE  1024
#define NGRP  8
#define TOPKG 3
#define TOPK  6
#define ISH   2048                  // shared-expert intermediate size
#define TOTSLOT (M_TOK * TOPK)      // 12288 routed token-slots
#define LDP   40                    // padded LDS row stride (bf16 elems, 80B)

typedef __attribute__((ext_vector_type(16))) __bf16 v16bf;
typedef __attribute__((ext_vector_type(8)))  __bf16 v8bf;
typedef __attribute__((ext_vector_type(8)))  float  v8f;

__device__ __forceinline__ v8f wmma_bf16(v16bf a, v16bf b, v8f c) {
  return __builtin_amdgcn_wmma_f32_16x16x32_bf16(false, a, false, b, (short)0, c,
                                                 false, false);
}

__device__ __forceinline__ float silu_mul(float g, float u) {
  // g * sigmoid(g) * u with v_rcp_f32 instead of IEEE divide
  return g * __builtin_amdgcn_rcpf(1.0f + __expf(-g)) * u;
}

__device__ __forceinline__ void cvt4_store(__bf16* dst, float4 v) {
  dst[0] = (__bf16)v.x; dst[1] = (__bf16)v.y;
  dst[2] = (__bf16)v.z; dst[3] = (__bf16)v.w;
}

// Read one wave-lane's WMMA A/B fragment (v16bf) from a staged LDS row:
// bytes [2*kb, 2*kb+16) and [2*kb+32, 2*kb+48)  (kb in {0,8})
__device__ __forceinline__ v16bf lds_frag(const __bf16* row, int kb) {
  v8bf lo = *(const v8bf*)(row + kb);
  v8bf hi = *(const v8bf*)(row + kb + 16);
  return __builtin_shufflevector(lo, hi, 0, 1, 2, 3, 4, 5, 6, 7,
                                 8, 9, 10, 11, 12, 13, 14, 15);
}

// ---------------------------------------------------------------------------
// 1) Router: scores = softmax(x @ gate_w^T); grouped top-k -> dense combine[M,E]
// ---------------------------------------------------------------------------
__global__ __launch_bounds__(64)
void router_kernel(const float* __restrict__ x, const float* __restrict__ gw,
                   float* __restrict__ combine) {
  __shared__ float s[NEXP];
  __shared__ float cw[NEXP];
  __shared__ float red;
  const int m = blockIdx.x;
  const int t = threadIdx.x;                 // one thread per expert
  const float* xr = x + (size_t)m * HID;
  const float* gr = gw + (size_t)t * HID;
  float acc = 0.f;
  for (int h = 0; h < HID; h += 4) {
    float4 xv = *(const float4*)(xr + h);
    float4 gv = *(const float4*)(gr + h);
    acc += xv.x * gv.x + xv.y * gv.y + xv.z * gv.z + xv.w * gv.w;
  }
  s[t] = acc;
  __syncthreads();
  if (t == 0) { float mx = s[0]; for (int i = 1; i < NEXP; ++i) mx = fmaxf(mx, s[i]); red = mx; }
  __syncthreads();
  float ex = __expf(acc - red);
  s[t] = ex;
  __syncthreads();
  if (t == 0) { float sm = 0.f; for (int i = 0; i < NEXP; ++i) sm += s[i]; red = sm; }
  __syncthreads();
  float prob = ex * __builtin_amdgcn_rcpf(red);
  s[t] = prob;
  cw[t] = 0.f;
  __syncthreads();
  if (t == 0) {
    float gs[NGRP];
    for (int g = 0; g < NGRP; ++g) {
      float mx = s[g * 8];
      for (int j = 1; j < 8; ++j) mx = fmaxf(mx, s[g * 8 + j]);
      gs[g] = mx;
    }
    bool gsel[NGRP] = {false, false, false, false, false, false, false, false};
    for (int j = 0; j < TOPKG; ++j) {
      int best = 0; float bv = -1e30f;
      for (int g = 0; g < NGRP; ++g)
        if (!gsel[g] && gs[g] > bv) { bv = gs[g]; best = g; }
      gsel[best] = true;
    }
    float tmp[NEXP];
    for (int e = 0; e < NEXP; ++e) tmp[e] = gsel[e >> 3] ? s[e] : 0.f;
    for (int j = 0; j < TOPK; ++j) {
      int best = 0; float bv = -1e30f;
      for (int e = 0; e < NEXP; ++e)
        if (tmp[e] > bv) { bv = tmp[e]; best = e; }
      cw[best] += bv;          // RENORMALIZE = False, ROUTED_SCALING = 1.0
      tmp[best] = -1e30f;
    }
  }
  __syncthreads();
  combine[(size_t)m * NEXP + t] = cw[t];
}

// ---------------------------------------------------------------------------
// 2) Per-expert token list build (one wave32 per expert, ordered compaction)
// ---------------------------------------------------------------------------
__global__ __launch_bounds__(32)
void build_lists(const float* __restrict__ combine, int* __restrict__ tok,
                 int* __restrict__ cnt) {
  const int e = blockIdx.x;
  const int lane = threadIdx.x;
  int c = 0;
  for (int base = 0; base < M_TOK; base += 32) {
    int m = base + lane;
    bool p = combine[(size_t)m * NEXP + e] != 0.f;
    unsigned mask = (unsigned)__ballot(p);
    if (p) tok[e * M_TOK + c + __popc(mask & ((1u << lane) - 1u))] = m;
    c += __popc(mask);
  }
  if (lane == 0) cnt[e] = c;
}

__global__ void calc_offsets(const int* __restrict__ cnt, int* __restrict__ off) {
  if (blockIdx.x == 0 && threadIdx.x == 0) {
    int o = 0;
    for (int e = 0; e < NEXP; ++e) { off[e] = o; o += cnt[e]; }
  }
}

// ---------------------------------------------------------------------------
// 3) Shared expert GEMM1 + silu*mul:  hs[M, ISH]
//    LDS-staged bf16 tiles: convert once, WMMA fragments via ds_load_b128.
// ---------------------------------------------------------------------------
__global__ __launch_bounds__(256)
void shared_gemm1_silu(const float* __restrict__ x, const float* __restrict__ sgu,
                       float* __restrict__ hs) {
  __shared__ __align__(16) __bf16 As[32 * LDP];
  __shared__ __align__(16) __bf16 Bgs[64 * LDP];
  __shared__ __align__(16) __bf16 Bus[64 * LDP];
  const int t  = threadIdx.x;
  const int mt = blockIdx.x;
  const int nb = blockIdx.y * 64;

  // cooperative fill assignment: A 4 floats/thread, each B 8 floats/thread
  const int ar = t >> 3, ac = (t & 7) * 4;           // A: 32 rows x 32
  const int br = t >> 2, bc = (t & 3) * 8;           // B: 64 rows x 32
  const float* ag = x   + (size_t)(mt * 32 + ar) * HID + ac;
  const float* gg = sgu + (size_t)(nb + br) * HID + bc;
  const float* ug = sgu + (size_t)(ISH + nb + br) * HID + bc;

  const int wid = t >> 5, lane = t & 31;
  const int wm = wid >> 2, wn = wid & 3;
  const int r = lane & 15, kb = (lane >> 4) << 3;
  const __bf16* aRow = As  + (wm * 16 + r) * LDP;
  const __bf16* gRow = Bgs + (wn * 16 + r) * LDP;
  const __bf16* uRow = Bus + (wn * 16 + r) * LDP;

  v8f accg = {0.f, 0.f, 0.f, 0.f, 0.f, 0.f, 0.f, 0.f};
  v8f accu = accg;
  for (int k = 0; k < HID; k += 32) {
    float4 av = *(const float4*)(ag + k);
    float4 g0 = *(const float4*)(gg + k);
    float4 g1 = *(const float4*)(gg + k + 4);
    float4 u0 = *(const float4*)(ug + k);
    float4 u1 = *(const float4*)(ug + k + 4);
    cvt4_store(As + ar * LDP + ac, av);
    cvt4_store(Bgs + br * LDP + bc, g0);
    cvt4_store(Bgs + br * LDP + bc + 4, g1);
    cvt4_store(Bus + br * LDP + bc, u0);
    cvt4_store(Bus + br * LDP + bc + 4, u1);
    __syncthreads();
    v16bf a  = lds_frag(aRow, kb);
    v16bf bg = lds_frag(gRow, kb);
    v16bf bu = lds_frag(uRow, kb);
    accg = wmma_bf16(a, bg, accg);
    accu = wmma_bf16(a, bu, accu);
    __syncthreads();
  }
  const int col = nb + wn * 16 + (lane & 15);
  const int rh  = (lane >> 4) * 8;
  const int m0  = mt * 32 + wm * 16;
#pragma unroll
  for (int v = 0; v < 8; ++v)
    hs[(size_t)(m0 + rh + v) * ISH + col] = silu_mul(accg[v], accu[v]);
}

// ---------------------------------------------------------------------------
// 4) Shared expert GEMM2: out[M,H] = hs @ shared_down^T  (first writer of out)
// ---------------------------------------------------------------------------
__global__ __launch_bounds__(256)
void shared_gemm2(const float* __restrict__ hs, const float* __restrict__ sdn,
                  float* __restrict__ out) {
  __shared__ __align__(16) __bf16 As[32 * LDP];
  __shared__ __align__(16) __bf16 Bs[64 * LDP];
  const int t  = threadIdx.x;
  const int mt = blockIdx.x;
  const int hb = blockIdx.y * 64;

  const int ar = t >> 3, ac = (t & 7) * 4;
  const int br = t >> 2, bc = (t & 3) * 8;
  const float* ag = hs  + (size_t)(mt * 32 + ar) * ISH + ac;
  const float* bg = sdn + (size_t)(hb + br) * ISH + bc;

  const int wid = t >> 5, lane = t & 31;
  const int wm = wid >> 2, wn = wid & 3;
  const int r = lane & 15, kb = (lane >> 4) << 3;
  const __bf16* aRow = As + (wm * 16 + r) * LDP;
  const __bf16* bRow = Bs + (wn * 16 + r) * LDP;

  v8f acc = {0.f, 0.f, 0.f, 0.f, 0.f, 0.f, 0.f, 0.f};
  for (int k = 0; k < ISH; k += 32) {
    float4 av = *(const float4*)(ag + k);
    float4 b0 = *(const float4*)(bg + k);
    float4 b1 = *(const float4*)(bg + k + 4);
    cvt4_store(As + ar * LDP + ac, av);
    cvt4_store(Bs + br * LDP + bc, b0);
    cvt4_store(Bs + br * LDP + bc + 4, b1);
    __syncthreads();
    v16bf a = lds_frag(aRow, kb);
    v16bf b = lds_frag(bRow, kb);
    acc = wmma_bf16(a, b, acc);
    __syncthreads();
  }
  const int col = hb + wn * 16 + (lane & 15);
  const int rh  = (lane >> 4) * 8;
  const int m0  = mt * 32 + wm * 16;
#pragma unroll
  for (int v = 0; v < 8; ++v)
    out[(size_t)(m0 + rh + v) * HID + col] = acc[v];
}

// ---------------------------------------------------------------------------
// 5) Routed GEMM1 + silu*mul over gathered tokens:  hroute[slot, NMOE]
//    grid ((E * 64 m-tiles) expert-major, NMOE/64); empty tiles early-exit.
// ---------------------------------------------------------------------------
__global__ __launch_bounds__(256)
void routed_gemm1_silu(const float* __restrict__ x, const float* __restrict__ w1,
                       const int* __restrict__ tok, const int* __restrict__ cnt,
                       const int* __restrict__ off, float* __restrict__ hbuf) {
  const int e  = blockIdx.x >> 6;
  const int mt = blockIdx.x & 63;
  const int T  = cnt[e];
  if (mt * 32 >= T) return;
  __shared__ __align__(16) __bf16 As[32 * LDP];
  __shared__ __align__(16) __bf16 Bgs[64 * LDP];
  __shared__ __align__(16) __bf16 Bus[64 * LDP];
  const int t  = threadIdx.x;
  const int nb = blockIdx.y * 64;

  // gathered A row for this thread's fill slice
  const int ar = t >> 3, ac = (t & 7) * 4;
  const int ia = mt * 32 + ar;
  const int token = tok[e * M_TOK + (ia < T ? ia : T - 1)];
  const float* ag = x + (size_t)token * HID + ac;

  const int br = t >> 2, bc = (t & 3) * 8;
  const float* gg = w1 + ((size_t)e * 2 * NMOE + nb + br) * HID + bc;
  const float* ug = w1 + ((size_t)e * 2 * NMOE + NMOE + nb + br) * HID + bc;

  const int wid = t >> 5, lane = t & 31;
  const int wm = wid >> 2, wn = wid & 3;
  const int r = lane & 15, kb = (lane >> 4) << 3;
  const __bf16* aRow = As  + (wm * 16 + r) * LDP;
  const __bf16* gRow = Bgs + (wn * 16 + r) * LDP;
  const __bf16* uRow = Bus + (wn * 16 + r) * LDP;

  v8f accg = {0.f, 0.f, 0.f, 0.f, 0.f, 0.f, 0.f, 0.f};
  v8f accu = accg;
  for (int k = 0; k < HID; k += 32) {
    float4 av = *(const float4*)(ag + k);
    float4 g0 = *(const float4*)(gg + k);
    float4 g1 = *(const float4*)(gg + k + 4);
    float4 u0 = *(const float4*)(ug + k);
    float4 u1 = *(const float4*)(ug + k + 4);
    cvt4_store(As + ar * LDP + ac, av);
    cvt4_store(Bgs + br * LDP + bc, g0);
    cvt4_store(Bgs + br * LDP + bc + 4, g1);
    cvt4_store(Bus + br * LDP + bc, u0);
    cvt4_store(Bus + br * LDP + bc + 4, u1);
    __syncthreads();
    v16bf a  = lds_frag(aRow, kb);
    v16bf bg = lds_frag(gRow, kb);
    v16bf bu = lds_frag(uRow, kb);
    accg = wmma_bf16(a, bg, accg);
    accu = wmma_bf16(a, bu, accu);
    __syncthreads();
  }
  const int col  = nb + wn * 16 + (lane & 15);
  const int rh   = (lane >> 4) * 8;
  const int base = off[e];
#pragma unroll
  for (int v = 0; v < 8; ++v) {
    int irow = mt * 32 + wm * 16 + rh + v;
    if (irow < T)
      hbuf[(size_t)(base + irow) * NMOE + col] = silu_mul(accg[v], accu[v]);
  }
}

// ---------------------------------------------------------------------------
// 6) Routed GEMM2 + weighted scatter-add:  out[m,:] += c * h_slot @ w2[e]^T
// ---------------------------------------------------------------------------
__global__ __launch_bounds__(256)
void routed_gemm2_scatter(const float* __restrict__ hbuf, const float* __restrict__ w2,
                          const int* __restrict__ tok, const int* __restrict__ cnt,
                          const int* __restrict__ off, const float* __restrict__ combine,
                          float* __restrict__ out) {
  const int e  = blockIdx.x >> 6;
  const int mt = blockIdx.x & 63;
  const int T  = cnt[e];
  if (mt * 32 >= T) return;
  __shared__ __align__(16) __bf16 As[32 * LDP];
  __shared__ __align__(16) __bf16 Bs[64 * LDP];
  __shared__ int   s_tok[32];
  __shared__ float s_w[32];
  const int t  = threadIdx.x;
  const int hb = blockIdx.y * 64;

  if (t < 32) {
    int i  = mt * 32 + t;
    int ic = i < T ? i : T - 1;
    int tkn = tok[e * M_TOK + ic];
    s_tok[t] = tkn;
    s_w[t]   = combine[(size_t)tkn * NEXP + e];
  }

  const int ar = t >> 3, ac = (t & 7) * 4;
  const int ia = mt * 32 + ar;
  const int sa = off[e] + (ia < T ? ia : T - 1);
  const float* ag = hbuf + (size_t)sa * NMOE + ac;

  const int br = t >> 2, bc = (t & 3) * 8;
  const float* bg = w2 + ((size_t)e * HID + hb + br) * NMOE + bc;

  const int wid = t >> 5, lane = t & 31;
  const int wm = wid >> 2, wn = wid & 3;
  const int r = lane & 15, kb = (lane >> 4) << 3;
  const __bf16* aRow = As + (wm * 16 + r) * LDP;
  const __bf16* bRow = Bs + (wn * 16 + r) * LDP;

  v8f acc = {0.f, 0.f, 0.f, 0.f, 0.f, 0.f, 0.f, 0.f};
  for (int k = 0; k < NMOE; k += 32) {
    float4 av = *(const float4*)(ag + k);
    float4 b0 = *(const float4*)(bg + k);
    float4 b1 = *(const float4*)(bg + k + 4);
    cvt4_store(As + ar * LDP + ac, av);
    cvt4_store(Bs + br * LDP + bc, b0);
    cvt4_store(Bs + br * LDP + bc + 4, b1);
    __syncthreads();
    v16bf a = lds_frag(aRow, kb);
    v16bf b = lds_frag(bRow, kb);
    acc = wmma_bf16(a, b, acc);
    __syncthreads();
  }
  const int col = hb + wn * 16 + (lane & 15);
  const int rh  = (lane >> 4) * 8;
#pragma unroll
  for (int v = 0; v < 8; ++v) {
    int rit  = wm * 16 + rh + v;          // row within 32-token tile
    int irow = mt * 32 + rit;
    if (irow < T)
      atomicAdd(out + (size_t)s_tok[rit] * HID + col, s_w[rit] * acc[v]);
  }
}

// ---------------------------------------------------------------------------
extern "C" void kernel_launch(void* const* d_in, const int* in_sizes, int n_in,
                              void* d_out, int out_size, void* d_ws, size_t ws_size,
                              hipStream_t stream) {
  (void)in_sizes; (void)n_in; (void)out_size; (void)ws_size;
  const float* x   = (const float*)d_in[0];
  const float* gw  = (const float*)d_in[1];
  const float* w1  = (const float*)d_in[2];
  const float* w2  = (const float*)d_in[3];
  const float* sgu = (const float*)d_in[4];
  const float* sdn = (const float*)d_in[5];
  float* out = (float*)d_out;

  char* ws = (char*)d_ws;
  size_t o = 0;
  float* combine = (float*)(ws + o); o += (size_t)M_TOK * NEXP * sizeof(float);
  int*   tok     = (int*)(ws + o);   o += (size_t)NEXP * M_TOK * sizeof(int);
  int*   cnt     = (int*)(ws + o);   o += 256;
  int*   off     = (int*)(ws + o);   o += 256;
  o = (o + 255) & ~(size_t)255;
  float* hs      = (float*)(ws + o); o += (size_t)M_TOK * ISH * sizeof(float);    // 16 MB
  float* hroute  = (float*)(ws + o); o += (size_t)TOTSLOT * NMOE * sizeof(float); // 48 MB

  router_kernel<<<M_TOK, 64, 0, stream>>>(x, gw, combine);
  build_lists<<<NEXP, 32, 0, stream>>>(combine, tok, cnt);
  calc_offsets<<<1, 1, 0, stream>>>(cnt, off);

  shared_gemm1_silu<<<dim3(M_TOK / 32, ISH / 64), 256, 0, stream>>>(x, sgu, hs);
  shared_gemm2<<<dim3(M_TOK / 32, HID / 64), 256, 0, stream>>>(hs, sdn, out);

  routed_gemm1_silu<<<dim3(NEXP * 64, NMOE / 64), 256, 0, stream>>>(x, w1, tok, cnt, off, hroute);
  routed_gemm2_scatter<<<dim3(NEXP * 64, HID / 64), 256, 0, stream>>>(hroute, w2, tok, cnt, off,
                                                                      combine, out);
}
